// SphericalHarmonicsEncoder_50302656970915
// MI455X (gfx1250) — matrix operand order repaired and checked by reading, not compiled, not run on hardware
//
#include <hip/hip_runtime.h>

// ---------------------------------------------------------------------------
// Fused SH(order16) encoder + 2-layer MLP (LN -> GELU -> LN) for gfx1250.
// Strategy: one wave32 owns a 16-ray tile. Basis staged in LDS as f16,
// GEMMs run on v_wmma_f32_16x16x32_f16, LayerNorms done cross-lane with
// shfl_xor inside the 16-lane half-groups of the WMMA C/D layout.
// Weights are pre-transposed to f16 [N][K] in workspace so each B fragment
// is one contiguous 32B global load (L2-resident, ~0.1 MB total weights).
// ---------------------------------------------------------------------------

typedef _Float16 v16h __attribute__((ext_vector_type(16)));
typedef _Float16 v8h  __attribute__((ext_vector_type(8)));
typedef float    v8f  __attribute__((ext_vector_type(8)));

#define NRAYS   262144
#define NC      289     // SH coeffs, order 16
#define NCP     320     // K padded to 10 * 32
#define HID     128
#define WAVES   4
#define RPW     16      // rays per wave
#define RPB     (WAVES * RPW)
#define LN_EPS  1e-5f

// ---- prep: W1[289x128] f32 -> W1T[128][320] f16 (zero pad); W2 likewise ----
__global__ void prep_weights_kernel(const float* __restrict__ W1,
                                    const float* __restrict__ W2,
                                    _Float16* __restrict__ W1T,
                                    _Float16* __restrict__ W2T) {
    const int stride = gridDim.x * blockDim.x;
    int tid = blockIdx.x * blockDim.x + threadIdx.x;
    for (int i = tid; i < HID * NCP; i += stride) {
        int n = i / NCP, k = i % NCP;
        W1T[i] = (k < NC) ? (_Float16)W1[k * HID + n] : (_Float16)0.0f;
    }
    for (int i = tid; i < HID * HID; i += stride) {
        int n = i / HID, k = i % HID;
        W2T[i] = (_Float16)W2[k * HID + n];
    }
}

__device__ __forceinline__ v16h cat8(v8h lo, v8h hi) {
    return __builtin_shufflevector(lo, hi, 0, 1, 2, 3, 4, 5, 6, 7,
                                           8, 9, 10, 11, 12, 13, 14, 15);
}

__launch_bounds__(WAVES * 32, 1)
__global__ void sh_mlp_fused_kernel(const float* __restrict__ dirs,
                                    const float* __restrict__ b1,
                                    const float* __restrict__ g1,
                                    const float* __restrict__ be1,
                                    const float* __restrict__ b2,
                                    const float* __restrict__ g2,
                                    const float* __restrict__ be2,
                                    const _Float16* __restrict__ W1T,
                                    const _Float16* __restrict__ W2T,
                                    float* __restrict__ out) {
    __shared__ __attribute__((aligned(32))) _Float16 sBuf[WAVES][RPW][NCP];

    const int tid  = threadIdx.x;
    const int wave = tid >> 5;
    const int lane = tid & 31;
    const int lrow = lane & 15;   // row (A) or col (B/C) within tile
    const int lhi  = lane >> 4;   // half-group selector
    const int rayBase = blockIdx.x * RPB + wave * RPW;

    // ---------------- Phase 1: SH basis -> LDS f16 [16][320] ----------------
    if (lane < 16) {
        const int ray = rayBase + lane;
        float dx = dirs[ray * 3 + 0] + 1e-8f;
        float dy = dirs[ray * 3 + 1] + 1e-8f;
        float dz = dirs[ray * 3 + 2] + 1e-8f;
        float nrm = sqrtf(dx * dx + dy * dy + dz * dz);
        float inv = 1.0f / fmaxf(nrm, 1e-8f);
        float x = fminf(fmaxf(dx * inv, -1.0f), 1.0f);
        float y = fminf(fmaxf(dy * inv, -1.0f), 1.0f);
        float z = fminf(fmaxf(dz * inv, -1.0f), 1.0f);
        float zc    = fminf(fmaxf(z, -0.9999999f), 0.9999999f);
        float cos_t = zc;                                   // cos(acos(zc))
        float sin_t = sqrtf(fmaxf(1.0f - zc * zc, 0.0f));   // theta in [0,pi]
        float phi = atan2f(y, x);
        float cp = cosf(phi), sp = sinf(phi);

        _Float16* row = &sBuf[wave][lane][0];
        int idx = 0;
        float cosl = 1.0f;
        for (int l = 0; l <= 16; ++l) {
            if (l == 0) { row[idx++] = (_Float16)1.0f; }
            else        { cosl *= cos_t; row[idx++] = (_Float16)cosl; }
            float norm = sqrtf(2.0f / (float)(2 * l + 1));
            float spw = sin_t;
            float cm = cp, sm = sp;               // cos(m*phi), sin(m*phi)
            for (int m = 1; m <= l; ++m) {
                row[idx++] = (_Float16)(norm * spw * cm);
                row[idx++] = (_Float16)(norm * spw * sm);
                spw *= sin_t;
                float cn = cm * cp - sm * sp;
                sm = sm * cp + cm * sp;
                cm = cn;
            }
        }
        for (int k = NC; k < NCP; ++k) row[k] = (_Float16)0.0f;
    }
    __syncthreads();

    // ---------------- Phase 2: GEMM1  [16x320] x [320x128] ------------------
    v8f acc[8];
#pragma unroll
    for (int t = 0; t < 8; ++t)
        acc[t] = (v8f){0.f, 0.f, 0.f, 0.f, 0.f, 0.f, 0.f, 0.f};

#pragma unroll
    for (int ks = 0; ks < NCP / 32; ++ks) {
        const int kb = ks * 32 + lhi * 16;       // A-frag K base for this lane
        v8h alo = *(const v8h*)&sBuf[wave][lrow][kb];
        v8h ahi = *(const v8h*)&sBuf[wave][lrow][kb + 8];
        v16h a = cat8(alo, ahi);
#pragma unroll
        for (int t = 0; t < 8; ++t) {
            const _Float16* wp = W1T + (t * 16 + lrow) * NCP + ks * 32 + lhi * 16;
            v16h b = *(const v16h*)wp;           // 16 contiguous K halves
            acc[t] = __builtin_amdgcn_wmma_f32_16x16x32_f16(
                false, a, false, b, (short)0, acc[t], false, false);
        }
    }
#pragma unroll
    for (int t = 0; t < 8; ++t) {
        float bb = b1[t * 16 + lrow];
#pragma unroll
        for (int r = 0; r < 8; ++r) acc[t][r] += bb;
    }

    // ------------- Phase 3: LayerNorm1 + exact GELU -> LDS f16 --------------
    _Float16* sH = &sBuf[wave][0][0];            // reuse as [16][128]
#pragma unroll
    for (int r = 0; r < 8; ++r) {
        float s = 0.0f, q = 0.0f;
#pragma unroll
        for (int t = 0; t < 8; ++t) { float v = acc[t][r]; s += v; q += v * v; }
#pragma unroll
        for (int m = 1; m <= 8; m <<= 1) {       // stays inside 16-lane group
            s += __shfl_xor(s, m, 32);
            q += __shfl_xor(q, m, 32);
        }
        float mu   = s * (1.0f / 128.0f);
        float var  = q * (1.0f / 128.0f) - mu * mu;
        float rstd = rsqrtf(var + LN_EPS);
        const int M = r + lhi * 8;               // C/D layout row
#pragma unroll
        for (int t = 0; t < 8; ++t) {
            const int n = t * 16 + lrow;
            float v = (acc[t][r] - mu) * rstd * g1[n] + be1[n];
            v = 0.5f * v * (1.0f + erff(v * 0.70710678118654752f));
            sH[M * HID + n] = (_Float16)v;
        }
    }
    __syncthreads();

    // ---------------- Phase 4: GEMM2  [16x128] x [128x128] ------------------
    v8f acc2[8];
#pragma unroll
    for (int t = 0; t < 8; ++t)
        acc2[t] = (v8f){0.f, 0.f, 0.f, 0.f, 0.f, 0.f, 0.f, 0.f};

#pragma unroll
    for (int ks = 0; ks < HID / 32; ++ks) {
        const int kb = ks * 32 + lhi * 16;
        v8h alo = *(const v8h*)&sH[lrow * HID + kb];
        v8h ahi = *(const v8h*)&sH[lrow * HID + kb + 8];
        v16h a = cat8(alo, ahi);
#pragma unroll
        for (int t = 0; t < 8; ++t) {
            const _Float16* wp = W2T + (t * 16 + lrow) * HID + ks * 32 + lhi * 16;
            v16h b = *(const v16h*)wp;
            acc2[t] = __builtin_amdgcn_wmma_f32_16x16x32_f16(
                false, a, false, b, (short)0, acc2[t], false, false);
        }
    }
#pragma unroll
    for (int t = 0; t < 8; ++t) {
        float bb = b2[t * 16 + lrow];
#pragma unroll
        for (int r = 0; r < 8; ++r) acc2[t][r] += bb;
    }

    // ---------------- Phase 5: LayerNorm2 -> f32 output ---------------------
#pragma unroll
    for (int r = 0; r < 8; ++r) {
        float s = 0.0f, q = 0.0f;
#pragma unroll
        for (int t = 0; t < 8; ++t) { float v = acc2[t][r]; s += v; q += v * v; }
#pragma unroll
        for (int m = 1; m <= 8; m <<= 1) {
            s += __shfl_xor(s, m, 32);
            q += __shfl_xor(q, m, 32);
        }
        float mu   = s * (1.0f / 128.0f);
        float var  = q * (1.0f / 128.0f) - mu * mu;
        float rstd = rsqrtf(var + LN_EPS);
        const int M = r + lhi * 8;
        float* orow = out + (size_t)(rayBase + M) * HID;
#pragma unroll
        for (int t = 0; t < 8; ++t) {
            const int n = t * 16 + lrow;
            orow[n] = (acc2[t][r] - mu) * rstd * g2[n] + be2[n];
        }
    }
}

extern "C" void kernel_launch(void* const* d_in, const int* in_sizes, int n_in,
                              void* d_out, int out_size, void* d_ws, size_t ws_size,
                              hipStream_t stream) {
    const float* dirs = (const float*)d_in[0];
    const float* W1   = (const float*)d_in[1];
    const float* b1   = (const float*)d_in[2];
    const float* g1   = (const float*)d_in[3];
    const float* be1  = (const float*)d_in[4];
    const float* W2   = (const float*)d_in[5];
    const float* b2   = (const float*)d_in[6];
    const float* g2   = (const float*)d_in[7];
    const float* be2  = (const float*)d_in[8];
    float* out = (float*)d_out;

    _Float16* W1T = (_Float16*)d_ws;                         // 128*320*2 = 80 KiB
    _Float16* W2T = (_Float16*)((char*)d_ws + HID * NCP * 2); // +32 KiB

    prep_weights_kernel<<<64, 256, 0, stream>>>(W1, W2, W1T, W2T);
    sh_mlp_fused_kernel<<<NRAYS / RPB, WAVES * 32, 0, stream>>>(
        dirs, b1, g1, be1, b2, g2, be2, W1T, W2T, out);
}
